// RVQVae_5093831213430
// MI455X (gfx1250) — matrix operand order repaired and checked by reading, not compiled, not run on hardware
//
#include <hip/hip_runtime.h>
#include <hip/hip_bf16.h>

typedef __attribute__((ext_vector_type(16))) __bf16 v16bf;
typedef __attribute__((ext_vector_type(8)))  float  v8f;

union BFrag { v16bf v; uint4 q[2]; };

// ---------------------------------------------------------------------------
// pack f32 row [C] -> bf16 row [Cp] with zero channel padding, optional relu
// grid.x = number of rows (B*T)
// ---------------------------------------------------------------------------
__global__ __launch_bounds__(256)
void k_pack_x(__hip_bfloat16* __restrict__ dst, const float* __restrict__ src,
              int C, int Cp, int relu)
{
  const int r = blockIdx.x;
  const float* s = src + (size_t)r * C;
  __hip_bfloat16* d = dst + (size_t)r * Cp;
  for (int c = threadIdx.x; c < Cp; c += blockDim.x) {
    float v = (c < C) ? s[c] : 0.f;
    if (relu) v = fmaxf(v, 0.f);
    d[c] = __float2bfloat16(v);
  }
}

// pack weights f32 [Co, Ci, Kt] -> bf16 [Kt, Cip, N] (zero padded)
// grid.x = Kt*Cip rows
__global__ __launch_bounds__(256)
void k_pack_w(__hip_bfloat16* __restrict__ dst, const float* __restrict__ w,
              int Co, int Ci, int Kt, int Cip, int N)
{
  const int rowid = blockIdx.x;            // k*Cip + ci
  const int k = rowid / Cip;
  const int ci = rowid - k * Cip;
  __hip_bfloat16* d = dst + (size_t)rowid * N;
  if (ci < Ci) {
    const float* ws = w + (size_t)ci * Kt + k;
    for (int co = threadIdx.x; co < N; co += blockDim.x)
      d[co] = __float2bfloat16(co < Co ? ws[(size_t)co * Ci * Kt] : 0.f);
  } else {
    for (int co = threadIdx.x; co < N; co += blockDim.x)
      d[co] = __float2bfloat16(0.f);
  }
}

// codebooks f32 [6,512(code),512(dim)] -> bf16 cbT [6,512(dim),512(code)]
// grid.x = 6*512 rows (q, dim)
__global__ __launch_bounds__(256)
void k_pack_cbt(__hip_bfloat16* __restrict__ dst, const float* __restrict__ cb)
{
  const int q = blockIdx.x >> 9;
  const int c = blockIdx.x & 511;          // dim
  __hip_bfloat16* d = dst + (size_t)blockIdx.x * 512;
  const float* s = cb + (size_t)q * 512 * 512 + c;
  for (int j = threadIdx.x; j < 512; j += blockDim.x)   // code
    d[j] = __float2bfloat16(s[(size_t)j * 512]);
}

__global__ __launch_bounds__(256)
void k_zero(float* __restrict__ p, int n)
{
  for (int i = blockIdx.x * blockDim.x + threadIdx.x; i < n;
       i += gridDim.x * blockDim.x) p[i] = 0.f;
}

// per-code squared norms: cn[q*512+j] = sum_c cb[q][j][c]^2
__global__ __launch_bounds__(256)
void k_cbnorm(float* __restrict__ cn, const float* __restrict__ cb)
{
  __shared__ float sh[256];
  const float* row = cb + (size_t)blockIdx.x * 512;
  float s = 0.f;
  for (int c = threadIdx.x; c < 512; c += 256) { float v = row[c]; s += v * v; }
  sh[threadIdx.x] = s; __syncthreads();
  for (int st = 128; st > 0; st >>= 1) {
    if (threadIdx.x < st) sh[threadIdx.x] += sh[threadIdx.x + st];
    __syncthreads();
  }
  if (threadIdx.x == 0) cn[blockIdx.x] = sh[0];
}

// ---------------------------------------------------------------------------
// Implicit-im2col conv1d as WMMA GEMM.
//   Y[b, t, co] = bias[co] + sum_{k, ci} Wp[k, ci, co] * X[b, src(t,k), ci]
// X packed bf16 [B, Tmem, Cip]; Wp bf16 [Ktaps*Cip, N] row-major, N mult of 128.
// 128 threads = 4 waves (2 M x 2 N); block tile 64(M) x 128(N);
// wave tile 32(M) x 64(N) = 2x4 WMMA fragments (8 wmma / 4 b128 A loads + 64 16-bit B loads).
// Out-of-range taps read from `zpad` (>=512 zero bf16) via pointer select.
// ups=1: conv runs in 2x-upsampled coords, memory row = t_src >> 1.
// ---------------------------------------------------------------------------
__global__ __launch_bounds__(128)
void k_conv(float* __restrict__ Y,
            const __hip_bfloat16* __restrict__ Xp,
            const __hip_bfloat16* __restrict__ Wp,
            const float* __restrict__ bias,
            const float* __restrict__ Res,
            const __hip_bfloat16* __restrict__ zpad,
            int Tout, int Tmem, int Cip, int Co, int N,
            int Ktaps, int stride, int dil, int pad, int ups, int reluOut)
{
  const int lane = threadIdx.x & 31;
  const int wave = threadIdx.x >> 5;
  const int t0 = blockIdx.x * 64 + (wave >> 1) * 32;
  const int c0 = blockIdx.y * 128 + (wave & 1) * 64;
  const int b  = blockIdx.z;

  const int row  = lane & 15;           // A: M row within fragment
  const int koff = (lane >> 4) * 8;     // A: K sub-offset per ISA layout
  const int Tlim = ups ? (Tmem << 1) : Tmem;
  const __hip_bfloat16* Xb = Xp + (size_t)b * Tmem * Cip;

  v8f acc[2][4] = {};

  // per-lane weight pointer: rows (tap*Cip + ci + 8*(lane>>4) + k), col c0 + (lane&15)
  const __hip_bfloat16* wrow = Wp + (size_t)((lane >> 4) * 8) * N + c0 + (lane & 15);

  for (int tap = 0; tap < Ktaps; ++tap) {
    // A base pointers for this tap (validity & row fixed across ci)
    const __hip_bfloat16* pA[2];
#pragma unroll
    for (int mf = 0; mf < 2; ++mf) {
      int ts = (t0 + mf * 16 + row) * stride + tap * dil - pad;
      bool ok = (ts >= 0) && (ts < Tlim);
      int tm = ups ? (ts >> 1) : ts;
      pA[mf] = (ok ? (Xb + (size_t)tm * Cip) : zpad) + koff;
    }
    for (int ci = 0; ci < Cip; ci += 32) {
      BFrag a[2], bb[4];
      const uint4* p0 = (const uint4*)(pA[0] + ci);
      const uint4* p1 = (const uint4*)(pA[1] + ci);
      a[0].q[0] = p0[0]; a[0].q[1] = p0[2];   // K halves per ISA A layout
      a[1].q[0] = p1[0]; a[1].q[1] = p1[2];
      const uint4* pw4 = (const uint4*)wrow;
#pragma unroll
      for (int nf = 0; nf < 4; ++nf)
#pragma unroll
        for (int s = 0; s < 16; ++s)
          bb[nf].v[s] = __builtin_bit_cast(__bf16, ((const unsigned short*)wrow)[(size_t)((s & 7) + ((s >> 3) << 4)) * N + nf * 16]);
      wrow += (size_t)32 * N;
#pragma unroll
      for (int mf = 0; mf < 2; ++mf)
#pragma unroll
        for (int nf = 0; nf < 4; ++nf)
          acc[mf][nf] = __builtin_amdgcn_wmma_f32_16x16x32_bf16(
              false, a[mf].v, false, bb[nf].v, (short)0, acc[mf][nf], false, false);
    }
  }

  // D layout: lane holds column N = lane&15; VGPR v holds M = v + 8*(lane>=16)
  const int mrow = (lane >> 4) * 8;
  const int ncol = lane & 15;
#pragma unroll
  for (int mf = 0; mf < 2; ++mf)
#pragma unroll
    for (int nf = 0; nf < 4; ++nf) {
      int co = c0 + nf * 16 + ncol;
      if (co >= Co) continue;
      float bv = bias[co];
#pragma unroll
      for (int v = 0; v < 8; ++v) {
        int t = t0 + mf * 16 + mrow + v;
        size_t oi = ((size_t)b * Tout + t) * Co + co;
        float val = acc[mf][nf][v] + bv;
        if (Res) val += Res[oi];
        if (reluOut) val = fmaxf(val, 0.f);
        Y[oi] = val;
      }
    }
}

// ---------------------------------------------------------------------------
// VQ per-row update: argmin_j (||c_j||^2 - 2 S[m,j]) with first-min tie-break,
// residual update R -= xq, accumulate Q += xq, loss and histogram atomics.
// ---------------------------------------------------------------------------
__global__ __launch_bounds__(256)
void k_vq_update(float* __restrict__ R, float* __restrict__ Q,
                 const float* __restrict__ S, const float* __restrict__ cb,
                 const float* __restrict__ cnorm, float* __restrict__ hist,
                 float* __restrict__ lossAcc)
{
  __shared__ float sd[256];
  __shared__ int   sj[256];
  const int m = blockIdx.x, tid = threadIdx.x;
  const float* Srow = S + (size_t)m * 512;

  float d0 = cnorm[tid]       - 2.f * Srow[tid];
  float d1 = cnorm[tid + 256] - 2.f * Srow[tid + 256];
  float bd = d0; int bj = tid;
  if (d1 < bd) { bd = d1; bj = tid + 256; }
  sd[tid] = bd; sj[tid] = bj; __syncthreads();
  for (int s = 128; s > 0; s >>= 1) {
    if (tid < s) {
      float od = sd[tid + s]; int oj = sj[tid + s];
      if (od < sd[tid] || (od == sd[tid] && oj < sj[tid])) { sd[tid] = od; sj[tid] = oj; }
    }
    __syncthreads();
  }
  const int idx = sj[0];
  __syncthreads();

  const float* cv = cb + (size_t)idx * 512;
  float* Rr = R + (size_t)m * 512;
  float* Qr = Q + (size_t)m * 512;
  float e2 = 0.f;
  for (int c = tid; c < 512; c += 256) {
    float xq = cv[c];
    float e = Rr[c] - xq;
    Rr[c] = e;
    Qr[c] += xq;
    e2 += e * e;
  }
  sd[tid] = e2; __syncthreads();
  for (int s = 128; s > 0; s >>= 1) {
    if (tid < s) sd[tid] += sd[tid + s];
    __syncthreads();
  }
  if (tid == 0) { atomicAdd(lossAcc, sd[0]); atomicAdd(&hist[idx], 1.0f); }
}

__global__ __launch_bounds__(256)
void k_finalize(float* __restrict__ out2, const float* __restrict__ lossAcc,
                const float* __restrict__ hist)
{
  __shared__ float sh[256];
  const int tid = threadIdx.x;
  float perp = 0.f;
  for (int q = 0; q < 6; ++q) {
    float s = 0.f;
    for (int j = tid; j < 512; j += 256) {
      float pj = hist[q * 512 + j] * (1.0f / 4096.0f);
      s += pj * logf(pj + 1e-10f);
    }
    sh[tid] = s; __syncthreads();
    for (int st = 128; st > 0; st >>= 1) {
      if (tid < st) sh[tid] += sh[tid + st];
      __syncthreads();
    }
    if (tid == 0) perp += expf(-sh[0]);
    __syncthreads();
  }
  if (tid == 0) {
    float l = 0.f;
    for (int q = 0; q < 6; ++q) l += lossAcc[q];
    out2[0] = l / (6.0f * 4096.0f * 512.0f);
    out2[1] = perp / 6.0f;
  }
}

// ---------------------------------------------------------------------------
// Host orchestration
// ---------------------------------------------------------------------------
extern "C" void kernel_launch(void* const* d_in, const int* in_sizes, int n_in,
                              void* d_out, int out_size, void* d_ws, size_t ws_size,
                              hipStream_t stream)
{
  if (n_in < 96) return;
  const int B = 64;
  auto IN = [&](int i) { return (const float*)d_in[i]; };

  // workspace carve (bump allocator, 256B aligned)
  uintptr_t p0 = (uintptr_t)d_ws, p = p0;
  auto alloc = [&](size_t bytes) -> void* {
    p = (p + 255) & ~(uintptr_t)255; void* r = (void*)p; p += bytes; return r;
  };
  float* A0 = (float*)alloc((size_t)64 * 512 * 512 * 4);
  float* A1 = (float*)alloc((size_t)64 * 512 * 512 * 4);
  __hip_bfloat16* XP  = (__hip_bfloat16*)alloc((size_t)64 * 512 * 512 * 2);
  __hip_bfloat16* WP  = (__hip_bfloat16*)alloc((size_t)4 * 512 * 512 * 2);
  __hip_bfloat16* CBT = (__hip_bfloat16*)alloc((size_t)6 * 512 * 512 * 2);
  float* CN = (float*)alloc((size_t)6 * 512 * 4);
  float* Q  = (float*)alloc((size_t)(4096 * 512 + 512 + 8 + 6 * 512) * 4);
  float* ZB   = Q + 4096 * 512;   // 512 zero f32 = 1024 zero bf16 (bias + A pad)
  float* LACC = ZB + 512;         // per-quantizer loss acc (8)
  float* HIST = LACC + 8;         // 6*512 histogram (float counts)
  if ((size_t)(p - p0) > ws_size) return;
  const __hip_bfloat16* ZPAD = (const __hip_bfloat16*)ZB;

  // zero Q / zero-bias / zero-pad / loss / hist up-front (conv reads ZPAD)
  int nz = 4096 * 512 + 512 + 8 + 6 * 512;
  k_zero<<<dim3((nz + 255) / 256), dim3(256), 0, stream>>>(Q, nz);

  auto conv = [&](float* dst, const float* src, const float* wF, const float* bF,
                  int Tin, int Tout, int Ci, int Co, int Kt,
                  int stride, int dil, int pad, int ups, int inRelu, int outRelu,
                  const float* res) {
    int Cip = (Ci + 31) & ~31;
    int N   = (Co + 127) & ~127;
    k_pack_x<<<dim3(B * Tin), dim3(256), 0, stream>>>(XP, src, Ci, Cip, inRelu);
    k_pack_w<<<dim3(Kt * Cip), dim3(256), 0, stream>>>(WP, wF, Co, Ci, Kt, Cip, N);
    k_conv<<<dim3(Tout / 64, N / 128, B), dim3(128), 0, stream>>>(
        dst, XP, WP, bF, res, ZPAD, Tout, Tin, Cip, Co, N, Kt, stride, dil, pad, ups, outRelu);
  };

  static const int d3[3] = {1, 3, 9};

  // ---------------- encoder ----------------
  conv(A0, (const float*)d_in[0], IN(51), IN(50), 512, 512, 263, 512, 3, 1, 1, 1, 0, 0, 1, nullptr);
  float* cur = A0; float* alt = A1; int T = 512;
  for (int l = 0; l < 3; ++l) {
    int base = 54 + l * 14;                              // down_b, down_w, res...
    conv(alt, cur, IN(base + 1), IN(base), T, T / 2, 512, 512, 4, 2, 1, 1, 0, 0, 0, nullptr);
    { float* t_ = cur; cur = alt; alt = t_; } T /= 2;
    for (int r = 0; r < 3; ++r) {
      int rb = base + 2 + r * 4;                         // b1, b2, w1, w2
      int dl = d3[r];
      conv(alt, cur, IN(rb + 2), IN(rb),     T, T, 512, 512, 3, 1, dl, dl, 0, 1, 0, nullptr);
      conv(cur, alt, IN(rb + 3), IN(rb + 1), T, T, 512, 512, 1, 1, 1, 0, 0, 1, 0, cur);
    }
  }
  conv(alt, cur, IN(53), IN(52), T, T, 512, 512, 3, 1, 1, 1, 0, 0, 0, nullptr);
  float* Z = alt;      // z -> becomes the in-place residual
  float* Sc = cur;     // free buffer reused for VQ scores

  // ---------------- residual VQ ----------------
  const float* cbk = IN(1);
  k_pack_cbt<<<dim3(6 * 512), dim3(256), 0, stream>>>(CBT, cbk);
  k_cbnorm<<<dim3(6 * 512), dim3(256), 0, stream>>>(CN, cbk);
  for (int q = 0; q < 6; ++q) {
    k_pack_x<<<dim3(4096), dim3(256), 0, stream>>>(XP, Z, 512, 512, 0);
    k_conv<<<dim3(1, 4, 64), dim3(128), 0, stream>>>(
        Sc, XP, CBT + (size_t)q * 512 * 512, ZB, nullptr, ZPAD,
        64, 64, 512, 512, 512, 1, 1, 1, 0, 0, 0);
    k_vq_update<<<dim3(4096), dim3(256), 0, stream>>>(
        Z, Q, Sc, cbk + (size_t)q * 512 * 512, CN + q * 512, HIST + q * 512, LACC + q);
  }

  // ---------------- decoder ----------------
  T = 64;
  conv(A0, Q, IN(3), IN(2), T, T, 512, 512, 3, 1, 1, 1, 0, 0, 1, nullptr);
  cur = A0; alt = A1;
  for (int l = 0; l < 3; ++l) {
    int base = 8 + l * 14;                               // res..., up_b, up_w
    for (int r = 0; r < 3; ++r) {
      int rb = base + r * 4;
      int dl = d3[2 - r];                                // 9, 3, 1
      conv(alt, cur, IN(rb + 2), IN(rb),     T, T, 512, 512, 3, 1, dl, dl, 0, 1, 0, nullptr);
      conv(cur, alt, IN(rb + 3), IN(rb + 1), T, T, 512, 512, 1, 1, 1, 0, 0, 1, 0, cur);
    }
    conv(alt, cur, IN(base + 13), IN(base + 12), T, 2 * T, 512, 512, 3, 1, 1, 1, 1, 0, 0, nullptr);
    { float* t_ = cur; cur = alt; alt = t_; } T *= 2;
  }
  conv(alt, cur, IN(5), IN(4), T, T, 512, 512, 3, 1, 1, 1, 0, 0, 1, nullptr);
  { float* t_ = cur; cur = alt; alt = t_; }
  conv((float*)d_out, cur, IN(7), IN(6), T, T, 512, 263, 3, 1, 1, 1, 0, 0, 0, nullptr);

  k_finalize<<<dim3(1), dim3(256), 0, stream>>>(
      (float*)d_out + (size_t)64 * 512 * 263, LACC, HIST);
}
